// ProductAttention_70978629533850
// MI455X (gfx1250) — matrix operand order repaired and checked by reading, not compiled, not run on hardware
//
#include <hip/hip_runtime.h>

typedef _Float16 half_t;
typedef __attribute__((ext_vector_type(16))) _Float16 v16h;
typedef __attribute__((ext_vector_type(8)))  _Float16 h8;
typedef __attribute__((ext_vector_type(8)))  float    v8f;

namespace {
constexpr int Tn = 2, Hh = 384, Ww = 384, Cc = 128;
constexpr int HEADS = 4, WINS = 8, HD = 32, NTOK = 64;
constexpr int NWH = Hh / WINS, NWW = Ww / WINS;       // 48, 48
constexpr int NWIN = Tn * NWH * NWW;                  // 4608
constexpr int NPIX = Tn * Hh * Ww;                    // 294912
constexpr float QSCALE = 0.17677669529663687f;        // 32^-0.5
}

union V16 { v16h v; h8 h[2]; };

// ---- CDNA5 async global->LDS copy (ASYNCcnt-tracked), per CDNA5 ISA ----
__device__ __forceinline__ void async_b128_to_lds(unsigned lds_off, const half_t* g) {
  asm volatile("global_load_async_to_lds_b128 %0, %1, off"
               :: "v"(lds_off), "v"(g) : "memory");
}
__device__ __forceinline__ void wait_async0() {
  asm volatile("s_wait_asynccnt 0x0" ::: "memory");
}
__device__ __forceinline__ unsigned lds_addr_of(const void* p) {
  // LDS aperture hi-bits live in addr[63:32]; low 32 bits are the wave-relative
  // LDS byte offset expected by GLOBAL_LOAD_ASYNC_TO_LDS (ISA 10.2).
  return (unsigned)(unsigned long long)p;
}

// ---------------- weight f32 -> f16 conversion (q_pw, k_pw, v_pw, proj_w) ---
__global__ void cvt_w_kernel(const float* __restrict__ a, const float* __restrict__ b,
                             const float* __restrict__ c, const float* __restrict__ d,
                             half_t* __restrict__ o) {
  int i = blockIdx.x * blockDim.x + threadIdx.x;
  if (i < Cc * Cc) {
    o[i]               = (half_t)a[i];
    o[Cc * Cc + i]     = (half_t)b[i];
    o[2 * Cc * Cc + i] = (half_t)c[i];
    o[3 * Cc * Cc + i] = (half_t)d[i];
  }
}

// ---------------- depthwise 3x3 + bias + ReLU, f32 NHWC -> f16 [NPIX][C] ----
__global__ void dw_kernel(const float* __restrict__ vid, const float* __restrict__ dw,
                          const float* __restrict__ dwb, half_t* __restrict__ out) {
  int gid = blockIdx.x * blockDim.x + threadIdx.x;
  if (gid >= NPIX * Cc) return;
  int c = gid & (Cc - 1);
  int p = gid >> 7;
  int x = p % Ww;
  int y = (p / Ww) % Hh;
  int t = p / (Ww * Hh);
  const float* base = vid + (size_t)t * Hh * Ww * Cc + c;
  float acc = 0.f;
#pragma unroll
  for (int dy = 0; dy < 3; ++dy) {
    int yy = y + dy - 1;
    if (yy < 0 || yy >= Hh) continue;
#pragma unroll
    for (int dx = 0; dx < 3; ++dx) {
      int xx = x + dx - 1;
      if (xx < 0 || xx >= Ww) continue;
      acc += base[((size_t)yy * Ww + xx) * Cc] * dw[c * 9 + dy * 3 + dx];
    }
  }
  acc += dwb[c];
  acc = acc > 0.f ? acc : 0.f;
  out[(size_t)p * Cc + c] = (half_t)acc;
}

// ---------------- pointwise 1x1 GEMM: [NPIX][128] x W[128][128]^T -----------
// Weight staged to LDS once per block via async copies; inner loop is
// ds_load_b128 -> v_wmma. mode 0: [win][head][tok][hd] (q/k);
// mode 2: [win][head][hd][tok] (v, transposed).
__global__ __launch_bounds__(256) void pw_gemm_kernel(
    const half_t* __restrict__ A, const half_t* __restrict__ Wh,
    const float* __restrict__ bias, half_t* __restrict__ out,
    int mode, float scale) {
  __shared__ half_t sB[Cc * Cc];          // 32 KB
  {
    unsigned l = lds_addr_of(sB) + threadIdx.x * 16;
    const half_t* g = Wh + threadIdx.x * 8;
#pragma unroll
    for (int it = 0; it < 8; ++it) {      // 256 thr * 16B * 8 = 32 KB
      async_b128_to_lds(l, g);
      l += 4096; g += 2048;
    }
    wait_async0();
  }
  __syncthreads();

  const int lane = threadIdx.x & 31;
  const int wave = threadIdx.x >> 5;
  const int mtile = blockIdx.x * 8 + wave;
  const int mbase = mtile * 16;
  const int r  = lane & 15;
  const int kh = (lane >> 4) * 8;    // A-operand K interleave offset
  const int bk = (lane >> 4) * 16;   // unused; B loads use kh, same k order as A

  v8f acc[8] = {};
  const half_t* arow = A + (size_t)(mbase + r) * Cc;
#pragma unroll
  for (int ks = 0; ks < 4; ++ks) {
    V16 a;
    a.h[0] = *(const h8*)(arow + ks * 32 + kh);
    a.h[1] = *(const h8*)(arow + ks * 32 + kh + 16);
#pragma unroll
    for (int nt = 0; nt < 8; ++nt) {
      const half_t* brow = sB + (nt * 16 + r) * Cc + ks * 32;
      V16 b;
      b.h[0] = *(const h8*)(brow + kh);
      b.h[1] = *(const h8*)(brow + kh + 16);
      acc[nt] = __builtin_amdgcn_wmma_f32_16x16x32_f16(
          false, a.v, false, b.v, (short)0, acc[nt], false, false);
    }
  }
#pragma unroll
  for (int i = 0; i < 8; ++i) {
    int row = (lane >> 4) * 8 + i;
    int p = mbase + row;
    int x = p % Ww;
    int y = (p / Ww) % Hh;
    int t = p / (Ww * Hh);
    int win = (t * NWH + y / WINS) * NWW + x / WINS;
    int tok = (y % WINS) * WINS + (x % WINS);
#pragma unroll
    for (int nt = 0; nt < 8; ++nt) {
      int col = nt * 16 + (lane & 15);
      float v = (acc[nt][i] + bias[col]) * scale;
      int head = col >> 5, hd = col & 31;
      size_t idx;
      if (mode == 2)
        idx = (((size_t)win * HEADS + head) * HD + hd) * NTOK + tok;
      else
        idx = (((size_t)win * HEADS + head) * NTOK + tok) * HD + hd;
      out[idx] = (half_t)v;
    }
  }
}

// ---------------- windowed attention: one wave per (window, head) -----------
__global__ __launch_bounds__(32) void attn_kernel(
    const half_t* __restrict__ qbuf, const half_t* __restrict__ kbuf,
    const half_t* __restrict__ vbuf, const float* __restrict__ rpb,
    half_t* __restrict__ outbuf) {
  __shared__ float  S[NTOK][NTOK];        // 16 KB
  __shared__ half_t P[NTOK][NTOK];        // 8 KB
  __shared__ half_t sK[NTOK * HD];        // 4 KB
  __shared__ half_t sV[HD * NTOK];        // 4 KB
  const int wh = blockIdx.x;               // win*HEADS + head
  const int head = wh & (HEADS - 1);
  const size_t win = (size_t)(wh >> 2);
  const int lane = threadIdx.x;
  const int r  = lane & 15;
  const int kh = (lane >> 4) * 8;
  const int bk = (lane >> 4) * 16;
  const half_t* Q  = qbuf + (size_t)wh * NTOK * HD;
  const half_t* Km = kbuf + (size_t)wh * NTOK * HD;
  const half_t* Vt = vbuf + (size_t)wh * HD * NTOK;

  // Async-stage K and V^T tiles into LDS (read 4x each below).
  {
    unsigned lK = lds_addr_of(sK) + lane * 16;
    unsigned lV = lds_addr_of(sV) + lane * 16;
    const half_t* gK = Km + lane * 8;
    const half_t* gV = Vt + lane * 8;
#pragma unroll
    for (int it = 0; it < 8; ++it) {      // 32 lanes * 16B * 8 = 4 KB each
      async_b128_to_lds(lK, gK);
      async_b128_to_lds(lV, gV);
      lK += 512; lV += 512; gK += 256; gV += 256;
    }
    wait_async0();
  }
  __syncthreads();

  // S = Q K^T  (K-dim = hd = 32 -> single WMMA k-step per tile)
#pragma unroll
  for (int mt = 0; mt < 4; ++mt) {
    V16 a;
    const half_t* arow = Q + (mt * 16 + r) * HD;
    a.h[0] = *(const h8*)(arow + kh);
    a.h[1] = *(const h8*)(arow + kh + 16);
#pragma unroll
    for (int nt = 0; nt < 4; ++nt) {
      const half_t* brow = sK + (nt * 16 + r) * HD;
      V16 b;
      b.h[0] = *(const h8*)(brow + kh);
      b.h[1] = *(const h8*)(brow + kh + 16);
      v8f c = {};
      c = __builtin_amdgcn_wmma_f32_16x16x32_f16(
          false, a.v, false, b.v, (short)0, c, false, false);
#pragma unroll
      for (int i = 0; i < 8; ++i)
        S[mt * 16 + (lane >> 4) * 8 + i][nt * 16 + (lane & 15)] = c[i];
    }
  }
  __syncthreads();

  // Swin relative-position-biased softmax over each row (2 rows per lane)
#pragma unroll
  for (int rr = 0; rr < 2; ++rr) {
    int row = lane + rr * 32;
    int iy1 = row >> 3, ix1 = row & 7;
    float mx = -1e30f;
    for (int m = 0; m < NTOK; ++m) {
      int iy2 = m >> 3, ix2 = m & 7;
      int idx = (iy1 - iy2 + 7) * 15 + (ix1 - ix2 + 7);
      float v = S[row][m] + rpb[idx * HEADS + head];
      S[row][m] = v;
      mx = fmaxf(mx, v);
    }
    float sum = 0.f;
    for (int m = 0; m < NTOK; ++m) {
      float e = __expf(S[row][m] - mx);
      S[row][m] = e;
      sum += e;
    }
    float inv = 1.f / sum;
    for (int m = 0; m < NTOK; ++m)
      P[row][m] = (half_t)(S[row][m] * inv);
  }
  __syncthreads();

  // O = P V  -> outbuf [win][tok][head*32 + hd]
#pragma unroll
  for (int mt = 0; mt < 4; ++mt) {
#pragma unroll
    for (int nt = 0; nt < 2; ++nt) {
      v8f c = {};
#pragma unroll
      for (int ks = 0; ks < 2; ++ks) {
        V16 a, b;
        const half_t* arow = &P[mt * 16 + r][ks * 32];
        a.h[0] = *(const h8*)(arow + kh);
        a.h[1] = *(const h8*)(arow + kh + 16);
        const half_t* brow = sV + (nt * 16 + r) * NTOK + ks * 32;
        b.h[0] = *(const h8*)(brow + kh);
        b.h[1] = *(const h8*)(brow + kh + 16);
        c = __builtin_amdgcn_wmma_f32_16x16x32_f16(
            false, a.v, false, b.v, (short)0, c, false, false);
      }
#pragma unroll
      for (int i = 0; i < 8; ++i) {
        int tok = mt * 16 + (lane >> 4) * 8 + i;
        int col = head * HD + nt * 16 + (lane & 15);
        outbuf[(win * NTOK + tok) * Cc + col] = (half_t)c[i];
      }
    }
  }
}

// ---------------- output projection GEMM, f32 NHWC out + bias --------------
__global__ __launch_bounds__(256) void proj_gemm_kernel(
    const half_t* __restrict__ A, const half_t* __restrict__ Wh,
    const float* __restrict__ bias, float* __restrict__ out) {
  __shared__ half_t sB[Cc * Cc];
  {
    unsigned l = lds_addr_of(sB) + threadIdx.x * 16;
    const half_t* g = Wh + threadIdx.x * 8;
#pragma unroll
    for (int it = 0; it < 8; ++it) {
      async_b128_to_lds(l, g);
      l += 4096; g += 2048;
    }
    wait_async0();
  }
  __syncthreads();

  const int lane = threadIdx.x & 31;
  const int wave = threadIdx.x >> 5;
  const int mtile = blockIdx.x * 8 + wave;
  const int mbase = mtile * 16;
  const int r  = lane & 15;
  const int kh = (lane >> 4) * 8;
  const int bk = (lane >> 4) * 16;

  v8f acc[8] = {};
  const half_t* arow = A + (size_t)(mbase + r) * Cc;
#pragma unroll
  for (int ks = 0; ks < 4; ++ks) {
    V16 a;
    a.h[0] = *(const h8*)(arow + ks * 32 + kh);
    a.h[1] = *(const h8*)(arow + ks * 32 + kh + 16);
#pragma unroll
    for (int nt = 0; nt < 8; ++nt) {
      const half_t* brow = sB + (nt * 16 + r) * Cc + ks * 32;
      V16 b;
      b.h[0] = *(const h8*)(brow + kh);
      b.h[1] = *(const h8*)(brow + kh + 16);
      acc[nt] = __builtin_amdgcn_wmma_f32_16x16x32_f16(
          false, a.v, false, b.v, (short)0, acc[nt], false, false);
    }
  }
#pragma unroll
  for (int i = 0; i < 8; ++i) {
    int row = (lane >> 4) * 8 + i;
    int m = mbase + row;               // m = win*64 + tok
    int win = m >> 6, tok = m & 63;
    int t = win / (NWH * NWW);
    int rem = win % (NWH * NWW);
    int y = (rem / NWW) * WINS + (tok >> 3);
    int x = (rem % NWW) * WINS + (tok & 7);
    size_t obase = (((size_t)t * Hh + y) * Ww + x) * Cc;
#pragma unroll
    for (int nt = 0; nt < 8; ++nt) {
      int col = nt * 16 + (lane & 15);
      out[obase + col] = acc[nt][i] + bias[col];
    }
  }
}

extern "C" void kernel_launch(void* const* d_in, const int* in_sizes, int n_in,
                              void* d_out, int out_size, void* d_ws, size_t ws_size,
                              hipStream_t stream) {
  (void)in_sizes; (void)n_in; (void)out_size; (void)ws_size;
  const float* vid  = (const float*)d_in[0];
  const float* rpb  = (const float*)d_in[1];
  const float* q_dw  = (const float*)d_in[2];
  const float* q_dwb = (const float*)d_in[3];
  const float* q_pw  = (const float*)d_in[4];
  const float* q_pwb = (const float*)d_in[5];
  const float* k_dw  = (const float*)d_in[6];
  const float* k_dwb = (const float*)d_in[7];
  const float* k_pw  = (const float*)d_in[8];
  const float* k_pwb = (const float*)d_in[9];
  const float* v_dw  = (const float*)d_in[10];
  const float* v_dwb = (const float*)d_in[11];
  const float* v_pw  = (const float*)d_in[12];
  const float* v_pwb = (const float*)d_in[13];
  const float* proj_w = (const float*)d_in[14];
  const float* proj_b = (const float*)d_in[15];
  float* out = (float*)d_out;

  const size_t bufElems = (size_t)NPIX * Cc;
  half_t* dwbuf = (half_t*)d_ws;           // depthwise output; later aliased as attn out
  half_t* qbuf  = dwbuf + bufElems;        // [win][head][tok][hd]
  half_t* kbuf  = qbuf + bufElems;         // [win][head][tok][hd]
  half_t* vbuf  = kbuf + bufElems;         // [win][head][hd][tok] (transposed)
  half_t* w16   = vbuf + bufElems;         // 4 x 128x128 f16 weights
  half_t* wq = w16, *wk = w16 + Cc * Cc, *wv = w16 + 2 * Cc * Cc, *wp = w16 + 3 * Cc * Cc;
  half_t* attnout = dwbuf;                 // alias: dwbuf dead after v GEMM

  const int dwBlocks   = (NPIX * Cc) / 256;       // 147456
  const int gemmBlocks = (NPIX / 16) / 8;         // 2304 (8 waves x 16 rows each)

  cvt_w_kernel<<<(Cc * Cc + 255) / 256, 256, 0, stream>>>(q_pw, k_pw, v_pw, proj_w, w16);

  // Q branch (SCALE folded into pointwise output)
  dw_kernel<<<dwBlocks, 256, 0, stream>>>(vid, q_dw, q_dwb, dwbuf);
  pw_gemm_kernel<<<gemmBlocks, 256, 0, stream>>>(dwbuf, wq, q_pwb, qbuf, 0, QSCALE);
  // K branch
  dw_kernel<<<dwBlocks, 256, 0, stream>>>(vid, k_dw, k_dwb, dwbuf);
  pw_gemm_kernel<<<gemmBlocks, 256, 0, stream>>>(dwbuf, wk, k_pwb, kbuf, 0, 1.0f);
  // V branch (written transposed per head for contiguous WMMA B-operand loads)
  dw_kernel<<<dwBlocks, 256, 0, stream>>>(vid, v_dw, v_dwb, dwbuf);
  pw_gemm_kernel<<<gemmBlocks, 256, 0, stream>>>(dwbuf, wv, v_pwb, vbuf, 2, 1.0f);

  // Attention: one wave per (window, head)
  attn_kernel<<<NWIN * HEADS, 32, 0, stream>>>(qbuf, kbuf, vbuf, rpb, attnout);

  // Output projection -> f32 NHWC
  proj_gemm_kernel<<<gemmBlocks, 256, 0, stream>>>(attnout, wp, proj_b, out);
}
